// MultiHeadAttention_66340064854577
// MI455X (gfx1250) — matrix operand order repaired and checked by reading, compiled, not run on hardware
//
#include <hip/hip_runtime.h>

#define D_MODEL 1024
#define NHEAD 16
#define HEAD_DIM 64
#define SEQ 2048
#define BATCH 2

typedef __attribute__((ext_vector_type(16))) __bf16 v16bf;
typedef __attribute__((ext_vector_type(8)))  float  v8f;
typedef __attribute__((ext_vector_type(4)))  unsigned int v4u;
typedef __attribute__((ext_vector_type(8)))  unsigned int v8u;

struct __attribute__((aligned(16))) U128 { unsigned int a, b, c, d; };
struct __attribute__((aligned(8)))  U64  { unsigned int x, y; };

union FragU { U128 u[2]; v16bf v; };

__device__ __forceinline__ unsigned short f2bf(float f) {
  unsigned int u = __float_as_uint(f);
  u += 0x7FFFu + ((u >> 16) & 1u);   // round-to-nearest-even
  return (unsigned short)(u >> 16);
}

__device__ __forceinline__ U64 pack4bf(float x, float y, float z, float w) {
  U64 r;
  r.x = (unsigned)f2bf(x) | ((unsigned)f2bf(y) << 16);
  r.y = (unsigned)f2bf(z) | ((unsigned)f2bf(w) << 16);
  return r;
}

// ---------------------------------------------------------------------------
// Tensor Data Mover: DMA a 64x64 bf16 tile (row stride 64 elems in memory)
// into LDS rows padded to 72 elems (pad 4 DWORDs every 32 DWORDs).
// 2D descriptor: groups 0/1 only (VADDR2/3 omitted -> NULL).
// ---------------------------------------------------------------------------
__device__ __forceinline__ void tdm_load_64x64_bf16(const unsigned short* gsrc,
                                                    unsigned lds_addr) {
  unsigned long long ga = (unsigned long long)(uintptr_t)gsrc;
  v4u g0;
  g0[0] = 1u;                                    // count=1, user mode
  g0[1] = lds_addr;                              // LDS byte address
  g0[2] = (unsigned)(ga & 0xFFFFFFFFu);          // global_addr[31:0]
  g0[3] = (unsigned)(ga >> 32) | 0x80000000u;    // global_addr[56:32] | type=2
  v8u g1;
  g1[0] = (1u << 16)                             // data_size = 2 bytes
        | (1u << 20)                             // pad_enable
        | (4u << 22)                             // pad_interval: 32 DWORDs
        | (3u << 25);                            // pad_amount: 4 DWORDs
  g1[1] = (64u << 16);                           // tensor_dim0 = 64
  g1[2] = (64u << 16);                           // tensor_dim1 = 64 (== tile)
  g1[3] = (64u << 16);                           // tile_dim0 = 64
  g1[4] = 64u;                                   // tile_dim1 = 64, tile_dim2=0
  g1[5] = 64u;                                   // tensor_dim0_stride = 64
  g1[6] = 0u;
  g1[7] = 0u;
  asm volatile("tensor_load_to_lds %0, %1" :: "s"(g0), "s"(g1) : "memory");
}

// ---------------------------------------------------------------------------
// Kernel 1: fused Q/K/V projections.  O[s,o] = sum_k X[s,k] * W[o,k]
// Block = 128 threads (4 waves), 64x64 output tile, K-step 32, bf16 WMMA.
// Output: bf16 in [B, H, S, 64] layout (Q pre-scaled by 1/sqrt(Dh)).
// ---------------------------------------------------------------------------
__global__ __launch_bounds__(128) void mha_proj_kernel(
    const float* __restrict__ query, const float* __restrict__ key,
    const float* __restrict__ value, const float* __restrict__ wq,
    const float* __restrict__ wk, const float* __restrict__ wv,
    unsigned short* __restrict__ qh, unsigned short* __restrict__ kh,
    unsigned short* __restrict__ vh) {
  __shared__ unsigned short As[64][40];   // 64 rows x 32 k (bf16), pad to 40
  __shared__ unsigned short Bs[64][40];

  const float* X; const float* W; unsigned short* dst; float scale;
  if (blockIdx.z == 0)      { X = query; W = wq; dst = qh; scale = 0.125f; }
  else if (blockIdx.z == 1) { X = key;   W = wk; dst = kh; scale = 1.0f; }
  else                      { X = value; W = wv; dst = vh; scale = 1.0f; }

  const int tid  = threadIdx.x;
  const int w    = tid >> 5;
  const int l    = tid & 31;
  const int half = l >> 4;
  const int ln   = l & 15;

  const int row0 = blockIdx.y * 64;   // M = B*S = 4096
  const int col0 = blockIdx.x * 64;   // N = 1024

  v8f acc[4];
  for (int g = 0; g < 4; ++g)
    for (int i = 0; i < 8; ++i) acc[g][i] = 0.0f;

  for (int k0 = 0; k0 < D_MODEL; k0 += 32) {
    __syncthreads();
#pragma unroll
    for (int p = 0; p < 4; ++p) {
      int r  = p * 16 + (tid >> 3);
      int c4 = (tid & 7) * 4;
      float4 xv = *(const float4*)&X[(size_t)(row0 + r) * D_MODEL + k0 + c4];
      *(U64*)&As[r][c4] = pack4bf(xv.x, xv.y, xv.z, xv.w);
      float4 wv4 = *(const float4*)&W[(size_t)(col0 + r) * D_MODEL + k0 + c4];
      *(U64*)&Bs[r][c4] = pack4bf(wv4.x, wv4.y, wv4.z, wv4.w);
    }
    __syncthreads();

    FragU a;   // A frag: row = w*16+ln, K chunks {half*8, 16+half*8}
    const unsigned short* ar = &As[w * 16 + ln][half * 8];
    a.u[0] = *(const U128*)ar;
    a.u[1] = *(const U128*)(ar + 16);
    FragU bb[4];
#pragma unroll
    for (int g = 0; g < 4; ++g) {   // batch all B-frag LDS loads first
      const unsigned short* br = &Bs[g * 16 + ln][half * 8];
      bb[g].u[0] = *(const U128*)br;
      bb[g].u[1] = *(const U128*)(br + 16);
    }
#pragma unroll
    for (int g = 0; g < 4; ++g)
      acc[g] = __builtin_amdgcn_wmma_f32_16x16x32_bf16(
          false, a.v, false, bb[g].v, (short)0, acc[g], false, false);
  }

  const int h = col0 >> 6;
  const int b = row0 >> 11;   // 2048 rows per batch, tiles never straddle
#pragma unroll
  for (int g = 0; g < 4; ++g) {
#pragma unroll
    for (int r = 0; r < 8; ++r) {
      int row = row0 + w * 16 + half * 8 + r;
      int s   = row & (SEQ - 1);
      int d   = g * 16 + ln;
      dst[((size_t)(b * NHEAD + h) * SEQ + s) * HEAD_DIM + d] =
          f2bf(acc[g][r] * scale);
    }
  }
}

// ---------------------------------------------------------------------------
// Kernel 2: flash attention.  One block per (b*H+h, 64 query rows).
// Q/K tiles streamed into LDS by the Tensor Data Mover; scores never
// touch HBM; online softmax; all matmuls via bf16 WMMA.
// ---------------------------------------------------------------------------
__global__ __launch_bounds__(128) void mha_attn_kernel(
    const unsigned short* __restrict__ qh, const unsigned short* __restrict__ kh,
    const unsigned short* __restrict__ vh, const int* __restrict__ mask,
    unsigned short* __restrict__ ctx) {
  __shared__ unsigned short Qs[64][72];
  __shared__ unsigned short Ks[64][72];
  __shared__ unsigned short Vt[64][72];   // V transposed: Vt[d][k]
  __shared__ unsigned short Ps[64][72];   // P (wave-private 16-row strips)

  const int tid  = threadIdx.x;
  const int w    = tid >> 5;
  const int l    = tid & 31;
  const int half = l >> 4;
  const int ln   = l & 15;

  const int q0 = blockIdx.x * 64;
  const int bh = blockIdx.y;
  const int b  = bh >> 4;
  const int h  = bh & 15;

  const unsigned ldsQ = (unsigned)(uintptr_t)&Qs[0][0];
  const unsigned ldsK = (unsigned)(uintptr_t)&Ks[0][0];

  // Q tile via TDM (issued once, by one wave; completion covered by the
  // first s_wait_tensorcnt below since TDM ops are in-order per wave).
  const unsigned short* qbase = qh + ((size_t)bh * SEQ + q0) * HEAD_DIM;
  if (w == 0) tdm_load_64x64_bf16(qbase, ldsQ);

  float m_i[8], l_i[8];
  v8f oA[4];
#pragma unroll
  for (int r = 0; r < 8; ++r) { m_i[r] = -3.0e38f; l_i[r] = 0.0f; }
  for (int g = 0; g < 4; ++g)
    for (int i = 0; i < 8; ++i) oA[g][i] = 0.0f;

  for (int kb = 0; kb < SEQ / 64; ++kb) {
    const int k0 = kb * 64;
    __syncthreads();                       // previous Ks/Vt fully consumed
    if (w == 0)
      tdm_load_64x64_bf16(kh + ((size_t)bh * SEQ + k0) * HEAD_DIM, ldsK);

    // V tile: manual load + transpose into Vt (TDM cannot transpose)
    const unsigned short* vbase = vh + ((size_t)bh * SEQ + k0) * HEAD_DIM;
#pragma unroll
    for (int p = 0; p < 4; ++p) {
      int r  = p * 16 + (tid >> 3);        // key index
      int c8 = (tid & 7) * 8;              // d index
      U128 vv = *(const U128*)&vbase[r * HEAD_DIM + c8];
      const unsigned short* pe = (const unsigned short*)&vv;
#pragma unroll
      for (int j = 0; j < 8; ++j) Vt[c8 + j][r] = pe[j];
    }

    // prefetch next key-block's mask tile while this block computes
    if (kb + 1 < SEQ / 64) {
      const int* pm = mask + ((size_t)b * SEQ + q0 + (tid >> 1)) * SEQ +
                      (k0 + 64) + (tid & 1) * 32;
      __builtin_prefetch(pm, 0, 1);
    }

    if (w == 0) __builtin_amdgcn_s_wait_tensorcnt(0);
    __syncthreads();

    // S = Q @ K^T : per wave 16 q-rows x 64 keys
    v8f sA[4];
    for (int g = 0; g < 4; ++g)
      for (int i = 0; i < 8; ++i) sA[g][i] = 0.0f;
#pragma unroll
    for (int ks = 0; ks < 64; ks += 32) {
      FragU a;
      const unsigned short* ar = &Qs[w * 16 + ln][ks + half * 8];
      a.u[0] = *(const U128*)ar;
      a.u[1] = *(const U128*)(ar + 16);
      FragU bb[4];
#pragma unroll
      for (int g = 0; g < 4; ++g) {        // B column n = key row in Ks
        const unsigned short* br = &Ks[g * 16 + ln][ks + half * 8];
        bb[g].u[0] = *(const U128*)br;
        bb[g].u[1] = *(const U128*)(br + 16);
      }
#pragma unroll
      for (int g = 0; g < 4; ++g)
        sA[g] = __builtin_amdgcn_wmma_f32_16x16x32_bf16(
            false, a.v, false, bb[g].v, (short)0, sA[g], false, false);
    }

    // mask + online softmax: row r lives in VGPR index r across a 16-lane half
#pragma unroll
    for (int r = 0; r < 8; ++r) {
      int qrow = q0 + w * 16 + half * 8 + r;
      const int* mr = mask + ((size_t)b * SEQ + qrow) * SEQ + k0;
      float bmax = -3.0e38f;
#pragma unroll
      for (int g = 0; g < 4; ++g) {
        float sv = sA[g][r];
        if (mr[g * 16 + ln] == 0) sv = -1.0e20f;
        sA[g][r] = sv;
        bmax = fmaxf(bmax, sv);
      }
#pragma unroll
      for (int off = 1; off < 16; off <<= 1)
        bmax = fmaxf(bmax, __shfl_xor(bmax, off, 32));
      float mnew  = fmaxf(m_i[r], bmax);
      float alpha = __expf(m_i[r] - mnew);
      m_i[r] = mnew;
      float rs = 0.0f;
#pragma unroll
      for (int g = 0; g < 4; ++g) {
        float pv = __expf(sA[g][r] - mnew);
        sA[g][r] = pv;
        rs += pv;
      }
#pragma unroll
      for (int off = 1; off < 16; off <<= 1)
        rs += __shfl_xor(rs, off, 32);
      l_i[r] = l_i[r] * alpha + rs;
#pragma unroll
      for (int g = 0; g < 4; ++g) {
        oA[g][r] *= alpha;
        Ps[w * 16 + half * 8 + r][g * 16 + ln] = f2bf(sA[g][r]);  // D->A relayout
      }
    }

    // context += P @ V   (A = wave's own Ps strip, B = Vt columns)
#pragma unroll
    for (int ks = 0; ks < 64; ks += 32) {
      FragU a;
      const unsigned short* ar = &Ps[w * 16 + ln][ks + half * 8];
      a.u[0] = *(const U128*)ar;
      a.u[1] = *(const U128*)(ar + 16);
      FragU bb[4];
#pragma unroll
      for (int g = 0; g < 4; ++g) {        // B column n = d index in Vt
        const unsigned short* br = &Vt[g * 16 + ln][ks + half * 8];
        bb[g].u[0] = *(const U128*)br;
        bb[g].u[1] = *(const U128*)(br + 16);
      }
#pragma unroll
      for (int g = 0; g < 4; ++g)
        oA[g] = __builtin_amdgcn_wmma_f32_16x16x32_bf16(
            false, a.v, false, bb[g].v, (short)0, oA[g], false, false);
    }
  }

  // normalize and store context as bf16 in [B, S, H*Dh]
#pragma unroll
  for (int r = 0; r < 8; ++r) {
    float inv = 1.0f / l_i[r];
    int qrow  = q0 + w * 16 + half * 8 + r;
    size_t base = ((size_t)b * SEQ + qrow) * D_MODEL + h * HEAD_DIM;
#pragma unroll
    for (int g = 0; g < 4; ++g)
      ctx[base + g * 16 + ln] = f2bf(oA[g][r] * inv);
  }
}

// ---------------------------------------------------------------------------
// Kernel 3: output projection. out[s,o] = sum_k ctx[s,k] * Wo[o,k], fp32 out.
// ---------------------------------------------------------------------------
__global__ __launch_bounds__(128) void mha_outproj_kernel(
    const unsigned short* __restrict__ ctxA, const float* __restrict__ wo,
    float* __restrict__ out) {
  __shared__ unsigned short As[64][40];
  __shared__ unsigned short Bs[64][40];

  const int tid  = threadIdx.x;
  const int w    = tid >> 5;
  const int l    = tid & 31;
  const int half = l >> 4;
  const int ln   = l & 15;

  const int row0 = blockIdx.y * 64;
  const int col0 = blockIdx.x * 64;

  v8f acc[4];
  for (int g = 0; g < 4; ++g)
    for (int i = 0; i < 8; ++i) acc[g][i] = 0.0f;

  for (int k0 = 0; k0 < D_MODEL; k0 += 32) {
    __syncthreads();
#pragma unroll
    for (int p = 0; p < 2; ++p) {        // bf16 A tile: straight copy
      int r  = p * 32 + (tid >> 2);
      int c8 = (tid & 3) * 8;
      *(U128*)&As[r][c8] =
          *(const U128*)&ctxA[(size_t)(row0 + r) * D_MODEL + k0 + c8];
    }
#pragma unroll
    for (int p = 0; p < 4; ++p) {        // fp32 Wo tile -> bf16
      int r  = p * 16 + (tid >> 3);
      int c4 = (tid & 7) * 4;
      float4 wv4 = *(const float4*)&wo[(size_t)(col0 + r) * D_MODEL + k0 + c4];
      *(U64*)&Bs[r][c4] = pack4bf(wv4.x, wv4.y, wv4.z, wv4.w);
    }
    __syncthreads();

    FragU a;
    const unsigned short* ar = &As[w * 16 + ln][half * 8];
    a.u[0] = *(const U128*)ar;
    a.u[1] = *(const U128*)(ar + 16);
    FragU bb[4];
#pragma unroll
    for (int g = 0; g < 4; ++g) {
      const unsigned short* br = &Bs[g * 16 + ln][half * 8];
      bb[g].u[0] = *(const U128*)br;
      bb[g].u[1] = *(const U128*)(br + 16);
    }
#pragma unroll
    for (int g = 0; g < 4; ++g)
      acc[g] = __builtin_amdgcn_wmma_f32_16x16x32_bf16(
          false, a.v, false, bb[g].v, (short)0, acc[g], false, false);
  }

#pragma unroll
  for (int g = 0; g < 4; ++g)
#pragma unroll
    for (int r = 0; r < 8; ++r) {
      int row = row0 + w * 16 + half * 8 + r;
      out[(size_t)row * D_MODEL + col0 + g * 16 + ln] = acc[g][r];
    }
}

// ---------------------------------------------------------------------------
extern "C" void kernel_launch(void* const* d_in, const int* in_sizes, int n_in,
                              void* d_out, int out_size, void* d_ws, size_t ws_size,
                              hipStream_t stream) {
  (void)in_sizes; (void)n_in; (void)out_size; (void)ws_size;
  const float* query = (const float*)d_in[0];
  const float* key   = (const float*)d_in[1];
  const float* value = (const float*)d_in[2];
  const int*   mask  = (const int*)d_in[3];
  const float* wq    = (const float*)d_in[4];
  const float* wk    = (const float*)d_in[5];
  const float* wv    = (const float*)d_in[6];
  const float* wo    = (const float*)d_in[7];
  float* out = (float*)d_out;

  const size_t elems = (size_t)BATCH * SEQ * D_MODEL;   // 4,194,304
  unsigned short* qh  = (unsigned short*)d_ws;          // 8 MiB each
  unsigned short* kh  = qh + elems;
  unsigned short* vh  = kh + elems;
  unsigned short* ctx = vh + elems;                     // total 32 MiB

  dim3 gProj(D_MODEL / 64, (BATCH * SEQ) / 64, 3);
  mha_proj_kernel<<<gProj, 128, 0, stream>>>(query, key, value, wq, wk, wv,
                                             qh, kh, vh);

  dim3 gAttn(SEQ / 64, BATCH * NHEAD);
  mha_attn_kernel<<<gAttn, 128, 0, stream>>>(qh, kh, vh, mask, ctx);

  dim3 gOut(D_MODEL / 64, (BATCH * SEQ) / 64);
  mha_outproj_kernel<<<gOut, 128, 0, stream>>>(ctx, wo, out);
}
